// dMoE_2018634629283
// MI455X (gfx1250) — matrix operand order repaired and checked by reading, not compiled, not run on hardware
//
#include <hip/hip_runtime.h>
#include <math.h>

// dMoE forward for MI455X (gfx1250): router -> top2 -> expert-grouped bf16 WMMA
// GEMMs (f32 accum, double-buffered LDS, async global->LDS for bf16 tiles)
// -> gelu -> GEMM -> weighted atomic scatter-combine.

#define Bv 4
#define Sv 2048
#define Hd 2048
#define Id 4096
#define Ev 8
#define Kk 2
#define Tt (Bv * Sv)              // 8192 tokens
#define TK (Tt * Kk)              // 16384 dispatched rows
#define MAXROWS (TK + Ev * 128)   // padded grouped-GEMM capacity = 17408
#define MAXB (MAXROWS / 128)      // 136 row-blocks max
#define LDA 40                    // LDS row stride (bf16 elems), padded vs 32

typedef __bf16 bf16_t;
typedef __attribute__((ext_vector_type(16))) __bf16 v16bf;
typedef __attribute__((ext_vector_type(8)))  __bf16 v8bf;
typedef __attribute__((ext_vector_type(8)))  float  v8f;

union Frag16 { v16bf v; v8bf h[2]; };
union F4 { float4 v; float f[4]; };

__device__ __forceinline__ float gelu_tanh(float x) {
    float x3 = x * x * x;
    float t = tanhf(0.7978845608028654f * (x + 0.044715f * x3));
    return 0.5f * x * (1.0f + t);
}

// pack 8 floats -> 8 bf16 -> one ds_store_b128
__device__ __forceinline__ void pack8_store(bf16_t* dst, const float* s) {
    union { bf16_t b[8]; uint4 u; } p;
#pragma unroll
    for (int i = 0; i < 8; ++i) p.b[i] = (bf16_t)s[i];
    *(uint4*)dst = p.u;
}
// pack 4 floats -> 4 bf16 -> one ds_store_b64
__device__ __forceinline__ void pack4_store(bf16_t* dst, float a, float b, float c, float d) {
    union { bf16_t b[4]; uint2 u; } p;
    p.b[0] = (bf16_t)a; p.b[1] = (bf16_t)b; p.b[2] = (bf16_t)c; p.b[3] = (bf16_t)d;
    *(uint2*)dst = p.u;
}

// gfx1250 async global->LDS copy (16B per lane), tracked by ASYNCcnt
__device__ __forceinline__ void async_b128(unsigned lds_off, const void* g) {
    asm volatile("global_load_async_to_lds_b128 %0, %1, off"
                 :: "v"(lds_off), "v"((unsigned long long)(uintptr_t)g)
                 : "memory");
}

// ---------------------------------------------------------------- init
__global__ void moe_init(float* __restrict__ y, int* __restrict__ tok,
                         float* __restrict__ wsrt, int* __restrict__ counts,
                         int* __restrict__ cursors) {
    long i = (long)blockIdx.x * blockDim.x + threadIdx.x;
    if (i < (long)Tt * Hd) y[i] = 0.0f;
    if (i < MAXROWS) { tok[i] = -1; wsrt[i] = 0.0f; }
    if (i < Ev) { counts[i] = 0; cursors[i] = 0; }
}

// ---------------------------------------------------------------- router
__global__ void moe_router(const float* __restrict__ x, const float* __restrict__ rw,
                           int* __restrict__ eidx, float* __restrict__ ew,
                           int* __restrict__ counts) {
    int t = (int)((blockIdx.x * blockDim.x + threadIdx.x) >> 5);
    int lane = threadIdx.x & 31;
    if (t >= Tt) return;
    const float* xr = x + (long)t * Hd;

    float acc[Ev];
#pragma unroll
    for (int e = 0; e < Ev; ++e) acc[e] = 0.0f;
    for (int h = lane; h < Hd; h += 32) {
        float xv = xr[h];
#pragma unroll
        for (int e = 0; e < Ev; ++e) acc[e] += xv * rw[e * Hd + h];
    }
#pragma unroll
    for (int e = 0; e < Ev; ++e)
        for (int off = 16; off > 0; off >>= 1)
            acc[e] += __shfl_down(acc[e], off, 32);

    if (lane == 0) {
        float m = acc[0];
#pragma unroll
        for (int e = 1; e < Ev; ++e) m = fmaxf(m, acc[e]);
        float p[Ev], s = 0.0f;
#pragma unroll
        for (int e = 0; e < Ev; ++e) { p[e] = __expf(acc[e] - m); s += p[e]; }
        float inv = 1.0f / s;
#pragma unroll
        for (int e = 0; e < Ev; ++e) p[e] *= inv;

        int i0 = 0; float v0 = p[0];
#pragma unroll
        for (int e = 1; e < Ev; ++e) if (p[e] > v0) { v0 = p[e]; i0 = e; }
        int i1 = -1; float v1 = -1.0f;
#pragma unroll
        for (int e = 0; e < Ev; ++e) if (e != i0 && p[e] > v1) { v1 = p[e]; i1 = e; }

        eidx[t * 2 + 0] = i0;  ew[t * 2 + 0] = v0;
        eidx[t * 2 + 1] = i1;  ew[t * 2 + 1] = v1;
        atomicAdd(&counts[i0], 1);
        atomicAdd(&counts[i1], 1);
    }
}

// ---------------------------------------------------------------- plan
__global__ void moe_plan(const int* __restrict__ counts, int* __restrict__ padoff,
                         int* __restrict__ blk_e, int* __restrict__ blk_r,
                         int* __restrict__ nblocks, int* __restrict__ out_counts) {
    if (blockIdx.x != 0 || threadIdx.x != 0) return;
    int total = 0, nb = 0;
    for (int e = 0; e < Ev; ++e) {
        padoff[e] = total;
        int c = counts[e];
        int nbk = (c + 127) >> 7;
        for (int b = 0; b < nbk; ++b) { blk_e[nb] = e; blk_r[nb] = total + b * 128; ++nb; }
        total += nbk << 7;
        out_counts[e] = c;          // tokens_per_expert (int32 bits) at d_out tail
    }
    padoff[Ev] = total;
    *nblocks = nb;
}

// ---------------------------------------------------------------- scatter
__global__ void moe_scatter(const int* __restrict__ eidx, const float* __restrict__ ew,
                            const int* __restrict__ padoff, int* __restrict__ cursors,
                            int* __restrict__ tok, float* __restrict__ wsrt) {
    int i = blockIdx.x * blockDim.x + threadIdx.x;
    if (i >= TK) return;
    int e = eidx[i];
    int r = atomicAdd(&cursors[e], 1);
    int pos = padoff[e] + r;
    tok[pos] = i >> 1;
    wsrt[pos] = ew[i];
}

// ---------------------------------------------------------------- GEMM1
// h[row, n] = gelu( x[tok[row], :] @ w1[e][:, n] ).  f32 global -> bf16 LDS,
// register-staged double buffering, one barrier per K-step.
__global__ void __launch_bounds__(256, 2)
moe_gemm1(const float* __restrict__ x, const float* __restrict__ w1,
          const int* __restrict__ tok, const int* __restrict__ blk_e,
          const int* __restrict__ blk_r, const int* __restrict__ nblocks,
          bf16_t* __restrict__ hbuf) {
    int b = blockIdx.x;
    if (b >= *nblocks) return;
    int e = blk_e[b];
    int row0 = blk_r[b];
    int n0g = blockIdx.y * 128;

    __shared__ alignas(16) bf16_t As[2][128 * LDA];
    __shared__ alignas(16) bf16_t Bs[2][128 * LDA];

    int tid = threadIdx.x;
    int lane = tid & 31;
    int w = tid >> 5;
    int wm = w & 3, wn = w >> 2;          // 4x2 wave grid -> 32x64 per wave
    int r = lane & 15, hi = lane >> 4;

    // A loader: 2 threads per row, 16 f32 each
    int lrow = tid >> 1;
    int lks  = (tid & 1) * 16;
    int mytok = tok[row0 + lrow];
    const float* arow = (mytok >= 0) ? (x + (long)mytok * Hd) : (const float*)0;

    // B loader: 4k x 4n micro-tile per thread (packs columns into b64 stores)
    int bkq = (tid >> 5) * 4;             // k offset 0..28
    int bng = (tid & 31) * 4;             // n offset 0..124
    const float* wbase = w1 + (long)e * Hd * Id + n0g + bng;

    float ar[16];
    float br[4][4];                       // [k][n]

    auto load_regs = [&](int k0) {
        if (arow) {
#pragma unroll
            for (int q = 0; q < 4; ++q) {
                F4 f; f.v = *(const float4*)(arow + k0 + lks + q * 4);
#pragma unroll
                for (int c = 0; c < 4; ++c) ar[q * 4 + c] = f.f[c];
            }
        } else {
#pragma unroll
            for (int q = 0; q < 16; ++q) ar[q] = 0.0f;
        }
#pragma unroll
        for (int i = 0; i < 4; ++i) {
            F4 f; f.v = *(const float4*)(wbase + (long)(k0 + bkq + i) * Id);
#pragma unroll
            for (int c = 0; c < 4; ++c) br[i][c] = f.f[c];
        }
    };
    auto store_tile = [&](int buf) {
        pack8_store(&As[buf][lrow * LDA + lks + 0], ar + 0);
        pack8_store(&As[buf][lrow * LDA + lks + 8], ar + 8);
#pragma unroll
        for (int nn = 0; nn < 4; ++nn)
            pack4_store(&Bs[buf][(bng + nn) * LDA + bkq],
                        br[0][nn], br[1][nn], br[2][nn], br[3][nn]);
    };

    v8f acc[2][4] = {};

    load_regs(0);
    store_tile(0);
    const int NT = Hd / 32;
    for (int kt = 0; kt < NT; ++kt) {
        int cur = kt & 1;
        bool next = (kt + 1 < NT);
        __syncthreads();                  // buf `cur` fully staged by all waves
        if (next) load_regs((kt + 1) * 32);
        if (kt + 2 < NT) {
            if (arow) __builtin_prefetch(arow + (kt + 2) * 32 + lks, 0, 1);
            __builtin_prefetch(wbase + (long)((kt + 2) * 32 + bkq) * Id, 0, 1);
        }

        Frag16 af[2], bf[4];
#pragma unroll
        for (int i = 0; i < 2; ++i) {
            int m = wm * 32 + i * 16 + r;
            af[i].h[0] = *(const v8bf*)&As[cur][m * LDA + hi * 8];
            af[i].h[1] = *(const v8bf*)&As[cur][m * LDA + hi * 8 + 16];
        }
#pragma unroll
        for (int j = 0; j < 4; ++j) {
            int n = wn * 64 + j * 16 + r;
            bf[j].h[0] = *(const v8bf*)&Bs[cur][n * LDA + hi * 8];
            bf[j].h[1] = *(const v8bf*)&Bs[cur][n * LDA + hi * 8 + 16];
        }
#pragma unroll
        for (int i = 0; i < 2; ++i)
#pragma unroll
            for (int j = 0; j < 4; ++j)
                acc[i][j] = __builtin_amdgcn_wmma_f32_16x16x32_bf16(
                    false, af[i].v, false, bf[j].v, (short)0, acc[i][j], false, false);

        if (next) store_tile(cur ^ 1);    // after barrier: nobody reads alt anymore
    }

    // epilogue: gelu + bf16 store to h
#pragma unroll
    for (int i = 0; i < 2; ++i) {
#pragma unroll
        for (int j = 0; j < 4; ++j) {
            int mbase = wm * 32 + i * 16 + hi * 8;
            int n = n0g + wn * 64 + j * 16 + r;
#pragma unroll
            for (int q = 0; q < 8; ++q) {
                float v = gelu_tanh(acc[i][j][q]);
                hbuf[(long)(row0 + mbase + q) * Id + n] = (bf16_t)v;
            }
        }
    }
}

// ---------------------------------------------------------------- GEMM2
// y[tok[row], n] += w_sorted[row] * ( h[row, :] @ w2[e][:, n] )
// A tile already bf16 -> async global->LDS copies (ASYNCcnt pipeline).
__global__ void __launch_bounds__(256, 2)
moe_gemm2(const bf16_t* __restrict__ hbuf, const float* __restrict__ w2,
          const int* __restrict__ tok, const float* __restrict__ wsrt,
          const int* __restrict__ blk_e, const int* __restrict__ blk_r,
          const int* __restrict__ nblocks, float* __restrict__ y) {
    int b = blockIdx.x;
    if (b >= *nblocks) return;
    int e = blk_e[b];
    int row0 = blk_r[b];
    int n0g = blockIdx.y * 128;

    __shared__ alignas(16) bf16_t As[2][128 * LDA];
    __shared__ alignas(16) bf16_t Bs[2][128 * LDA];

    int tid = threadIdx.x;
    int lane = tid & 31;
    int w = tid >> 5;
    int wm = w & 3, wn = w >> 2;
    int r = lane & 15, hi = lane >> 4;

    int lrow = tid >> 1;
    int lks  = (tid & 1) * 16;
    const bf16_t* arow = hbuf + (long)(row0 + lrow) * Id;

    int bkq = (tid >> 5) * 4;
    int bng = (tid & 31) * 4;
    const float* wbase = w2 + (long)e * Id * Hd + n0g + bng;

    float br[4][4];

    auto issue_asyncA = [&](int k0, int buf) {        // 32B per thread, 2x b128
        unsigned off0 = (unsigned)(uintptr_t)&As[buf][lrow * LDA + lks];
        const bf16_t* g = arow + k0 + lks;
        async_b128(off0, g);
        async_b128(off0 + 16u, g + 8);
    };
    auto loadB_regs = [&](int k0) {
#pragma unroll
        for (int i = 0; i < 4; ++i) {
            F4 f; f.v = *(const float4*)(wbase + (long)(k0 + bkq + i) * Hd);
#pragma unroll
            for (int c = 0; c < 4; ++c) br[i][c] = f.f[c];
        }
    };
    auto storeB_tile = [&](int buf) {
#pragma unroll
        for (int nn = 0; nn < 4; ++nn)
            pack4_store(&Bs[buf][(bng + nn) * LDA + bkq],
                        br[0][nn], br[1][nn], br[2][nn], br[3][nn]);
    };

    v8f acc[2][4] = {};

    issue_asyncA(0, 0);
    loadB_regs(0);
    storeB_tile(0);
    const int NT = Id / 32;
    for (int kt = 0; kt < NT; ++kt) {
        int cur = kt & 1;
        bool next = (kt + 1 < NT);
        asm volatile("s_wait_asynccnt 0x0" ::: "memory");  // own As[cur] landed
        __syncthreads();                                   // everyone's tiles ready
        if (next) {
            issue_asyncA((kt + 1) * 32, cur ^ 1);          // alt free for all waves
            loadB_regs((kt + 1) * 32);
        }

        Frag16 af[2], bf[4];
#pragma unroll
        for (int i = 0; i < 2; ++i) {
            int m = wm * 32 + i * 16 + r;
            af[i].h[0] = *(const v8bf*)&As[cur][m * LDA + hi * 8];
            af[i].h[1] = *(const v8bf*)&As[cur][m * LDA + hi * 8 + 16];
        }
#pragma unroll
        for (int j = 0; j < 4; ++j) {
            int n = wn * 64 + j * 16 + r;
            bf[j].h[0] = *(const v8bf*)&Bs[cur][n * LDA + hi * 8];
            bf[j].h[1] = *(const v8bf*)&Bs[cur][n * LDA + hi * 8 + 16];
        }
#pragma unroll
        for (int i = 0; i < 2; ++i)
#pragma unroll
            for (int j = 0; j < 4; ++j)
                acc[i][j] = __builtin_amdgcn_wmma_f32_16x16x32_bf16(
                    false, af[i].v, false, bf[j].v, (short)0, acc[i][j], false, false);

        if (next) storeB_tile(cur ^ 1);
    }

    // epilogue: weighted scatter-add combine into y
#pragma unroll
    for (int i = 0; i < 2; ++i) {
#pragma unroll
        for (int j = 0; j < 4; ++j) {
            int mbase = wm * 32 + i * 16 + hi * 8;
            int n = n0g + wn * 64 + j * 16 + r;
#pragma unroll
            for (int q = 0; q < 8; ++q) {
                int row = row0 + mbase + q;
                int t = tok[row];
                if (t >= 0) {
                    float v = acc[i][j][q] * wsrt[row];
                    atomicAdd(&y[(long)t * Hd + n], v);
                }
            }
        }
    }
}

// ---------------------------------------------------------------- launch
extern "C" void kernel_launch(void* const* d_in, const int* in_sizes, int n_in,
                              void* d_out, int out_size, void* d_ws, size_t ws_size,
                              hipStream_t stream) {
    const float* x  = (const float*)d_in[0];
    const float* rw = (const float*)d_in[1];
    const float* w1 = (const float*)d_in[2];
    const float* w2 = (const float*)d_in[3];
    float* y = (float*)d_out;
    int* out_counts = (int*)((float*)d_out + (long)Tt * Hd);

    char* ws = (char*)d_ws;
    size_t off = 0;
    auto alloc = [&](size_t bytes) -> void* {
        void* p = ws + off;
        off = (off + bytes + 255) & ~(size_t)255;
        return p;
    };
    int*    eidx    = (int*)   alloc(sizeof(int) * TK);
    float*  ew      = (float*) alloc(sizeof(float) * TK);
    int*    counts  = (int*)   alloc(sizeof(int) * Ev);
    int*    cursors = (int*)   alloc(sizeof(int) * Ev);
    int*    padoff  = (int*)   alloc(sizeof(int) * (Ev + 1));
    int*    blk_e   = (int*)   alloc(sizeof(int) * MAXB);
    int*    blk_r   = (int*)   alloc(sizeof(int) * MAXB);
    int*    nblk    = (int*)   alloc(sizeof(int));
    int*    tok     = (int*)   alloc(sizeof(int) * MAXROWS);
    float*  wsrt    = (float*) alloc(sizeof(float) * MAXROWS);
    bf16_t* hbuf    = (bf16_t*)alloc(sizeof(bf16_t) * (size_t)MAXROWS * Id);

    moe_init<<<((long)Tt * Hd + 255) / 256, 256, 0, stream>>>(y, tok, wsrt, counts, cursors);
    moe_router<<<(Tt * 32) / 256, 256, 0, stream>>>(x, rw, eidx, ew, counts);
    moe_plan<<<1, 64, 0, stream>>>(counts, padoff, blk_e, blk_r, nblk, out_counts);
    moe_scatter<<<(TK + 255) / 256, 256, 0, stream>>>(eidx, ew, padoff, cursors, tok, wsrt);
    moe_gemm1<<<dim3(MAXB, Id / 128), 256, 0, stream>>>(x, w1, tok, blk_e, blk_r, nblk, hbuf);
    moe_gemm2<<<dim3(MAXB, Hd / 128), 256, 0, stream>>>(hbuf, w2, tok, wsrt, blk_e, blk_r, nblk, y);
}
